// OctonionCausalSelfAttention_12756052869299
// MI455X (gfx1250) — matrix operand order repaired and checked by reading, not compiled, not run on hardware
//
#include <hip/hip_runtime.h>

// ---------------------------------------------------------------------------
// Octonion causal self-attention for MI455X (gfx1250, wave32, WMMA).
// All GEMM-shaped work runs on v_wmma_f32_16x16x32_bf16. Every WMMA fragment
// is fetched as two B128 loads (buffers are pre-swizzled into fragment-tile
// order). GEMM B panels are staged through LDS via the gfx1250 async
// load-to-LDS path (global_load_async_to_lds_b128 + s_wait_asynccnt).
// ---------------------------------------------------------------------------

#define TSEQ  2048
#define CEMB  1024
#define NHEAD 16
#define HDIM  64

typedef __attribute__((ext_vector_type(16))) __bf16          v16bf;
typedef __attribute__((ext_vector_type(16))) unsigned short  v16u;
typedef __attribute__((ext_vector_type(8)))  float           v8f;
typedef __attribute__((ext_vector_type(4)))  unsigned int    v4u;
typedef __attribute__((ext_vector_type(4)))  int             v4i;

#define NEG_INF (-__builtin_inff())

#if __has_builtin(__builtin_amdgcn_global_load_async_to_lds_b128) && \
    __has_builtin(__builtin_amdgcn_s_wait_asynccnt)
#define HAVE_ASYNC_LDS 1
#endif
#define AS1 __attribute__((address_space(1)))
#define AS3 __attribute__((address_space(3)))

static __device__ __forceinline__ unsigned short f2bf(float f) {
  unsigned int u = __float_as_uint(f);
  u += 0x7fffu + ((u >> 16) & 1u);          // round-to-nearest-even
  return (unsigned short)(u >> 16);
}

static __device__ __forceinline__ v16u pack2(v4u lo, v4u hi) {
  union { v4u q[2]; v16u v; } u;
  u.q[0] = lo; u.q[1] = hi;
  return u.v;
}

static __device__ __forceinline__ v8f wmma_bf16(v16u a, v16u b, v8f c) {
  return __builtin_amdgcn_wmma_f32_16x16x32_bf16(
      false, __builtin_bit_cast(v16bf, a),
      false, __builtin_bit_cast(v16bf, b),
      (short)0, c, false, false);
}

// A fragment (16x32 bf16, MxK) from row-major memory: per-lane data is one
// contiguous 16-element run -> two B128 loads.
static __device__ __forceinline__ v16u load_a_frag(const unsigned short* A,
                                                   int lda, int m0, int k0,
                                                   int lane) {
  int m = m0 + (lane & 15);
  int half = lane >> 4;
  const unsigned short* p = A + (size_t)m * lda + k0 + half * 16;
  return pack2(*(const v4u*)p, *(const v4u*)(p + 8));
}

// B fragment where B = K^T (scores): element (kk,n) = krow[jbase+n][kk];
// per-lane data is 16 contiguous elements of one k-row -> two B128 loads.
static __device__ __forceinline__ v16u load_b_krows(const unsigned short* kbase,
                                                    int jbase, int k0,
                                                    int lane) {
  int j = jbase + (lane & 15);
  int half = lane >> 4;
  const unsigned short* p = kbase + (size_t)j * HDIM + k0 + half * 16;
  return pack2(*(const v4u*)p, *(const v4u*)(p + 8));
}

// B fragment from fragment-swizzled storage: tile = [32 lanes][16 ushorts].
static __device__ __forceinline__ v16u load_b_swz(const unsigned short* base,
                                                  size_t tile, int lane) {
  const unsigned short* p = base + (tile << 9) + lane * 16;
  return pack2(*(const v4u*)p, *(const v4u*)(p + 8));
}

// ------------------------- conversion kernels ------------------------------

__global__ void cvt_bf16(const float* __restrict__ src,
                         unsigned short* __restrict__ dst, int n) {
  int i = blockIdx.x * blockDim.x + threadIdx.x;
  if (i < n) dst[i] = f2bf(src[i]);
}

// W[N][K] f32 -> fragment-swizzled bf16: dst[((kt*(N/16)+nt)*32+lane)*16+s]
// holds W[nt*16 + lane%16][kt*32 + (lane/16)*16 + s].
__global__ void cvt_w_swz(const float* __restrict__ W,
                          unsigned short* __restrict__ dst, int N, int K) {
  int id = blockIdx.x * blockDim.x + threadIdx.x;
  if (id >= N * K) return;
  int s = id & 15;
  int rest = id >> 4;
  int lane = rest & 31;
  rest >>= 5;
  int nt = rest % (N / 16);
  int kt = rest / (N / 16);
  int n = nt * 16 + (lane & 15);
  int kk = kt * 32 + (lane >> 4) * 16 + s;
  dst[id] = f2bf(W[(size_t)n * K + kk]);
}

// ------------------------------- GEMM --------------------------------------
// C[M][N] f32 = A[M][K] bf16 @ B (fragment-swizzled) bf16.
// Block: 4 waves = 64 rows of M x 64 cols of N; B panel staged via LDS.

__global__ __launch_bounds__(128) void gemm_bf16(
    const unsigned short* __restrict__ A, const unsigned short* __restrict__ Bsw,
    float* __restrict__ C, int M, int N, int K) {
  __shared__ __align__(16) unsigned short Bs[2048];  // 4 KB: 4 n-tiles x 512
  int tid = threadIdx.x;
  int lane = tid & 31, wave = tid >> 5;
  int m0 = blockIdx.y * 64 + wave * 16;
  int n0 = blockIdx.x * 64;
  int ntn = N >> 4;
  v8f z = {0.f, 0.f, 0.f, 0.f, 0.f, 0.f, 0.f, 0.f};
  v8f acc[4] = {z, z, z, z};

  for (int k0 = 0; k0 < K; k0 += 32) {
    size_t tile0 = (size_t)(k0 >> 5) * ntn + (n0 >> 4);
    const unsigned short* src = Bsw + (tile0 << 9);
#ifdef HAVE_ASYNC_LDS
    {
      AS1 char* g = (AS1 char*)src;                // cast away const for builtin
      AS3 char* l = (AS3 char*)Bs;
      __builtin_amdgcn_global_load_async_to_lds_b128(
          (AS1 v4i*)(g + tid * 16), (AS3 v4i*)(l + tid * 16), 0, 0);
      __builtin_amdgcn_global_load_async_to_lds_b128(
          (AS1 v4i*)(g + 2048 + tid * 16), (AS3 v4i*)(l + 2048 + tid * 16), 0,
          0);
      __builtin_amdgcn_s_wait_asynccnt(0);
    }
#else
    *(v4u*)((char*)Bs + tid * 16) = *(const v4u*)((const char*)src + tid * 16);
    *(v4u*)((char*)Bs + 2048 + tid * 16) =
        *(const v4u*)((const char*)src + 2048 + tid * 16);
#endif
    __syncthreads();
    v16u a = load_a_frag(A, K, m0, k0, lane);
#pragma unroll
    for (int t = 0; t < 4; ++t) {
      const unsigned short* p = Bs + t * 512 + lane * 16;
      v16u b = pack2(*(const v4u*)p, *(const v4u*)(p + 8));
      acc[t] = wmma_bf16(a, b, acc[t]);
    }
    __syncthreads();
  }
  int half = lane >> 4, nl = lane & 15;
#pragma unroll
  for (int t = 0; t < 4; ++t)
#pragma unroll
    for (int r = 0; r < 8; ++r)
      C[(size_t)(m0 + r + 8 * half) * N + n0 + t * 16 + nl] = acc[t][r];
}

// --------------------- RoPE + RMSNorm + KV build ---------------------------
// KVflat is written directly in fragment-swizzled order:
// tile = (h*(T/32) + j/32)*32 + n/16 ; lane = ((j%32)/16)*16 + n%16 ; s = j%16

__global__ __launch_bounds__(256) void qkv_post(
    const float* __restrict__ q0, const float* __restrict__ k0,
    const float* __restrict__ v0, const float* __restrict__ cosb,
    const float* __restrict__ sinb, unsigned short* __restrict__ qb,
    unsigned short* __restrict__ kb, float* __restrict__ qf,
    unsigned short* __restrict__ kv) {
  int id = blockIdx.x * 256 + threadIdx.x;
  if (id >= NHEAD * TSEQ) return;
  int h = id / TSEQ, t = id % TSEQ;

  float qv[64], kw[64], vr[64];
#pragma unroll 8
  for (int d = 0; d < 64; ++d) {
    qv[d] = q0[(size_t)t * CEMB + h * 64 + d];
    kw[d] = k0[(size_t)t * CEMB + h * 64 + d];
    vr[d] = v0[(size_t)t * CEMB + h * 64 + d];
  }
  for (int which = 0; which < 2; ++which) {
    float* xp = which ? kw : qv;
    float y[64];
    for (int d = 0; d < 32; ++d) {
      float c = cosb[(size_t)t * 32 + d], s = sinb[(size_t)t * 32 + d];
      y[d]      = xp[d] * c + xp[d + 32] * s;
      y[d + 32] = xp[d + 32] * c - xp[d] * s;
    }
    float ss = 0.f;
    for (int d = 0; d < 64; ++d) ss += y[d] * y[d];
    float r = rsqrtf(ss * (1.f / 64.f) + 1e-6f);
    for (int d = 0; d < 64; ++d) xp[d] = y[d] * r;
  }
  size_t row = (size_t)h * TSEQ + t;
  for (int d = 0; d < 64; ++d) {
    qb[row * 64 + d] = f2bf(qv[d]);
    qf[row * 64 + d] = qv[d];
    kb[row * 64 + d] = f2bf(kw[d]);
  }
  // octonion-normalize k, conjugate, outer with v -> swizzled KVflat
  int jt = t >> 5;
  int jhalf = (t >> 4) & 1;
  int sjj = t & 15;
  size_t tbase = ((size_t)(h * (TSEQ / 32) + jt) * 32) << 9;  // tile nt=0 base
  for (int m = 0; m < 8; ++m) {
    float nrm = 0.f;
    for (int a = 0; a < 8; ++a) nrm += kw[m * 8 + a] * kw[m * 8 + a];
    nrm = fmaxf(sqrtf(nrm), 1e-12f);
    float kc[8];
    kc[0] = kw[m * 8] / nrm;
    for (int a = 1; a < 8; ++a) kc[a] = -kw[m * 8 + a] / nrm;
    for (int p = 0; p < 8; ++p)
      for (int q = 0; q < 8; ++q) {
        int n = m * 64 + p * 8 + q;
        size_t addr = tbase + ((size_t)(n >> 4) << 9) +
                      (size_t)(jhalf * 16 + (n & 15)) * 16 + sjj;
        kv[addr] = f2bf(kc[p] * vr[m * 8 + q]);
      }
  }
}

// ----------------------- octonion algebra (f32) ----------------------------

static __device__ __forceinline__ void qmulf(const float* a, const float* b,
                                             float* o) {
  o[0] = a[0] * b[0] - a[1] * b[1] - a[2] * b[2] - a[3] * b[3];
  o[1] = a[0] * b[1] + a[1] * b[0] + a[2] * b[3] - a[3] * b[2];
  o[2] = a[0] * b[2] - a[1] * b[3] + a[2] * b[0] + a[3] * b[1];
  o[3] = a[0] * b[3] + a[1] * b[2] - a[2] * b[1] + a[3] * b[0];
}

static __device__ __forceinline__ void omulf(const float* o1, const float* o2,
                                             float* out) {
  const float* a = o1;
  const float* b = o1 + 4;
  const float* c = o2;
  const float* d = o2 + 4;
  float dc[4] = {d[0], -d[1], -d[2], -d[3]};
  float cc[4] = {c[0], -c[1], -c[2], -c[3]};
  float t1[4], t2[4], t3[4], t4[4];
  qmulf(a, c, t1);
  qmulf(dc, b, t2);
  qmulf(d, a, t3);
  qmulf(b, cc, t4);
  for (int k = 0; k < 4; ++k) {
    out[k] = t1[k] - t2[k];
    out[4 + k] = t3[k] + t4[k];
  }
}

// ------------------- fused causal attention (per head, 16 rows) ------------

static __device__ __forceinline__ void score_chunk(
    int jc, int i0, int lane, int wave, v16u aq0, v16u aq1,
    const unsigned short* kbase, float (*Sch)[128]) {
  int jb = jc + wave * 16;
  v8f s = {0.f, 0.f, 0.f, 0.f, 0.f, 0.f, 0.f, 0.f};
  v16u b0 = load_b_krows(kbase, jb, 0, lane);
  v16u b1 = load_b_krows(kbase, jb, 32, lane);
  s = wmma_bf16(aq0, b0, s);
  s = wmma_bf16(aq1, b1, s);
  int nl = lane & 15, half = lane >> 4;
  int j = jb + nl;
#pragma unroll
  for (int r = 0; r < 8; ++r) {
    int i = i0 + r + 8 * half;
    Sch[r + 8 * half][wave * 16 + nl] = (j <= i) ? s[r] * 0.125f : NEG_INF;
  }
}

__global__ __launch_bounds__(256) void attn_fused(
    const unsigned short* __restrict__ qb, const unsigned short* __restrict__ kb,
    const unsigned short* __restrict__ kv, const float* __restrict__ qf,
    unsigned short* __restrict__ yb) {
  __shared__ __align__(16) float qfs[16][64];
  __shared__ __align__(16) float Sch[16][128];
  __shared__ __align__(16) unsigned short Pch[16][128];
  __shared__ float rowm[16], rowl[16];
  __shared__ __align__(16) float zbuf[16][512];

  int h = blockIdx.y;
  int i0 = blockIdx.x * 16;
  int tid = threadIdx.x;
  int lane = tid & 31, wave = tid >> 5;
  int nl = lane & 15, half = lane >> 4;

  for (int idx = tid; idx < 16 * 64; idx += 256)
    qfs[idx >> 6][idx & 63] =
        qf[((size_t)h * TSEQ + i0 + (idx >> 6)) * 64 + (idx & 63)];
  if (tid < 16) {
    rowm[tid] = NEG_INF;
    rowl[tid] = 0.f;
  }
  const unsigned short* qh = qb + (size_t)h * TSEQ * 64;
  const unsigned short* kh = kb + (size_t)h * TSEQ * 64;
  v16u aq0 = load_a_frag(qh, 64, i0, 0, lane);
  v16u aq1 = load_a_frag(qh, 64, i0, 32, lane);
  __syncthreads();

  int nj = i0 + 16;  // valid key columns: 0 .. i0+15

  // ---- pass A: exact streaming softmax stats -----------------------------
  for (int jc = 0; jc < nj; jc += 128) {
    score_chunk(jc, i0, lane, wave, aq0, aq1, kh, Sch);
    __syncthreads();
    if (tid < 16) {
      float cmax = NEG_INF;
      for (int c = 0; c < 128; ++c) cmax = fmaxf(cmax, Sch[tid][c]);
      if (cmax > NEG_INF) {
        float m_old = rowm[tid];
        float m_new = fmaxf(m_old, cmax);
        float l = rowl[tid] * __expf(m_old - m_new);
        for (int c = 0; c < 128; ++c) {
          float sv = Sch[tid][c];
          l += (sv > NEG_INF) ? __expf(sv - m_new) : 0.f;
        }
        rowm[tid] = m_new;
        rowl[tid] = l;
      }
    }
    __syncthreads();
  }

  // ---- pass B: probs (bf16) and Z = P @ KVflat ---------------------------
  v8f z8 = {0.f, 0.f, 0.f, 0.f, 0.f, 0.f, 0.f, 0.f};
  v8f zacc[4] = {z8, z8, z8, z8};
  for (int jc = 0; jc < nj; jc += 128) {
    score_chunk(jc, i0, lane, wave, aq0, aq1, kh, Sch);
    __syncthreads();
    for (int idx = tid; idx < 16 * 128; idx += 256) {
      int r = idx >> 7, c = idx & 127;
      float sv = Sch[r][c];
      Pch[r][c] = f2bf((sv > NEG_INF) ? __expf(sv - rowm[r]) : 0.f);
    }
    __syncthreads();
#pragma unroll
    for (int ks = 0; ks < 4; ++ks) {
      const unsigned short* pp = &Pch[nl][ks * 32 + half * 16];
      v16u ap = pack2(*(const v4u*)pp, *(const v4u*)(pp + 8));
      size_t jt = (size_t)(h * (TSEQ / 32) + ((jc + ks * 32) >> 5));
#pragma unroll
      for (int t = 0; t < 4; ++t) {
        v16u b = load_b_swz(kv, jt * 32 + wave * 4 + t, lane);
        zacc[t] = wmma_bf16(ap, b, zacc[t]);
      }
    }
    __syncthreads();
  }

  // normalize Z by row sums, stage in LDS
#pragma unroll
  for (int t = 0; t < 4; ++t)
#pragma unroll
    for (int r = 0; r < 8; ++r) {
      int m = r + 8 * half;
      zbuf[m][wave * 64 + t * 16 + nl] = zacc[t][r] / rowl[m];
    }
  __syncthreads();

  // ---- octonion contraction: y = sum_q omul(omul(q8, Z[:,q]), e_q) -------
  if (tid < 128) {
    int i = tid >> 3, m = tid & 7;
    float q8[8], col[8], u[8], t8[8], y[8];
    for (int a = 0; a < 8; ++a) q8[a] = qfs[i][m * 8 + a];
    for (int k = 0; k < 8; ++k) y[k] = 0.f;
    for (int q = 0; q < 8; ++q) {
      for (int p = 0; p < 8; ++p) col[p] = zbuf[i][m * 64 + p * 8 + q];
      omulf(q8, col, u);
      float e[8] = {0.f, 0.f, 0.f, 0.f, 0.f, 0.f, 0.f, 0.f};
      e[q] = 1.f;
      omulf(u, e, t8);
      for (int k = 0; k < 8; ++k) y[k] += t8[k];
    }
    for (int k = 0; k < 8; ++k)
      yb[(size_t)(i0 + i) * CEMB + h * 64 + m * 8 + k] = f2bf(y[k]);
  }
}

// ------------------------------ launch -------------------------------------

extern "C" void kernel_launch(void* const* d_in, const int* in_sizes, int n_in,
                              void* d_out, int out_size, void* d_ws,
                              size_t ws_size, hipStream_t stream) {
  const float* x    = (const float*)d_in[0];
  const float* cosb = (const float*)d_in[1];
  const float* sinb = (const float*)d_in[2];
  const float* Wq   = (const float*)d_in[3];
  const float* Wk   = (const float*)d_in[4];
  const float* Wv   = (const float*)d_in[5];
  const float* Wo   = (const float*)d_in[6];

  char* ws = (char*)d_ws;
  const size_t MB = 1024 * 1024;
  unsigned short* xb  = (unsigned short*)(ws + 0 * MB);   // 4 MB  x bf16
  unsigned short* wqS = (unsigned short*)(ws + 4 * MB);   // 2 MB  swizzled
  unsigned short* wkS = (unsigned short*)(ws + 6 * MB);   // 2 MB
  unsigned short* wvS = (unsigned short*)(ws + 8 * MB);   // 2 MB
  unsigned short* woS = (unsigned short*)(ws + 10 * MB);  // 2 MB
  float*          q0  = (float*)(ws + 12 * MB);           // 8 MB
  float*          k0  = (float*)(ws + 20 * MB);           // 8 MB
  float*          v0  = (float*)(ws + 28 * MB);           // 8 MB
  unsigned short* qb  = (unsigned short*)(ws + 36 * MB);  // 4 MB  [h][t][64]
  unsigned short* kb  = (unsigned short*)(ws + 40 * MB);  // 4 MB  [h][t][64]
  float*          qfp = (float*)(ws + 44 * MB);           // 8 MB  [h][t][64]
  unsigned short* kvb = (unsigned short*)(ws + 52 * MB);  // 32 MB swizzled KV
  unsigned short* yb  = (unsigned short*)(ws + 84 * MB);  // 4 MB  [t][1024]
  (void)in_sizes; (void)n_in; (void)out_size; (void)ws_size;

  const int NX = TSEQ * CEMB;
  const int NW = CEMB * CEMB;

  cvt_bf16<<<(NX + 255) / 256, 256, 0, stream>>>(x, xb, NX);
  cvt_w_swz<<<(NW + 255) / 256, 256, 0, stream>>>(Wq, wqS, CEMB, CEMB);
  cvt_w_swz<<<(NW + 255) / 256, 256, 0, stream>>>(Wk, wkS, CEMB, CEMB);
  cvt_w_swz<<<(NW + 255) / 256, 256, 0, stream>>>(Wv, wvS, CEMB, CEMB);
  cvt_w_swz<<<(NW + 255) / 256, 256, 0, stream>>>(Wo, woS, CEMB, CEMB);

  dim3 gg(CEMB / 64, TSEQ / 64);  // (16, 32), 4 waves per block
  gemm_bf16<<<gg, 128, 0, stream>>>(xb, wqS, q0, TSEQ, CEMB, CEMB);
  gemm_bf16<<<gg, 128, 0, stream>>>(xb, wkS, k0, TSEQ, CEMB, CEMB);
  gemm_bf16<<<gg, 128, 0, stream>>>(xb, wvS, v0, TSEQ, CEMB, CEMB);

  qkv_post<<<(NHEAD * TSEQ + 255) / 256, 256, 0, stream>>>(
      q0, k0, v0, cosb, sinb, qb, kb, qfp, kvb);

  attn_fused<<<dim3(TSEQ / 16, NHEAD), 256, 0, stream>>>(qb, kb, kvb, qfp, yb);

  gemm_bf16<<<gg, 128, 0, stream>>>(yb, woS, (float*)d_out, TSEQ, CEMB, CEMB);
}
